// ClusterMemory_7164005449845
// MI455X (gfx1250) — matrix operand order repaired and checked by reading, not compiled, not run on hardware
//
#include <hip/hip_runtime.h>
#include <math.h>

// ---------------- problem constants (match reference) ----------------
constexpr int   kB = 256;       // batch rows
constexpr int   kD = 2048;      // feature dim (K)
constexpr int   kN = 100000;    // memory bank columns (N)
constexpr float kTemp = 0.05f;
constexpr float kEps  = 1e-12f;

// ---------------- tiling ----------------
constexpr int BM = 128;         // rows per block (grid.y = 2)
constexpr int BN = 128;         // cols per block
constexpr int KC = 32;          // K chunk (bf16 WMMA depth)
constexpr int PITCH = 40;       // LDS row pitch in halfs (32 data + 8 pad, 80B: kills bank conflicts)

typedef __attribute__((ext_vector_type(16))) __bf16 v16bf;
typedef __attribute__((ext_vector_type(8)))  float  v8f;
typedef __attribute__((ext_vector_type(8)))  unsigned short us8;

union V16 { v16bf bf; us8 u[2]; };

__device__ __forceinline__ unsigned short f32_to_bf16_rne(float f) {
    unsigned int u = __float_as_uint(f);
    u += 0x7fffu + ((u >> 16) & 1u);
    return (unsigned short)(u >> 16);
}
__device__ __forceinline__ float bf16_to_f32(unsigned short h) {
    return __uint_as_float(((unsigned int)h) << 16);
}

// =====================================================================
// Kernel 1: per-row L2 norm -> pre-scaled xs (f32) + bf16 hi/lo split.
// Also zero-initializes rowsum (one row per block).
// =====================================================================
__global__ void prep_kernel(const float* __restrict__ inputs,
                            float* __restrict__ xs,
                            unsigned short* __restrict__ xhi,
                            unsigned short* __restrict__ xlo,
                            float* __restrict__ rowsum) {
    const int row = blockIdx.x;
    const int t   = threadIdx.x;
    __shared__ float red[256];

    float ss = 0.f;
    for (int k = t; k < kD; k += 256) {
        float v = inputs[row * kD + k];
        ss += v * v;
    }
    red[t] = ss;
    __syncthreads();
    for (int s = 128; s > 0; s >>= 1) {
        if (t < s) red[t] += red[t + s];
        __syncthreads();
    }
    const float scale = 1.0f / (fmaxf(sqrtf(red[0]), kEps) * kTemp);

    for (int k = t; k < kD; k += 256) {
        float v = inputs[row * kD + k] * scale;   // pre-scaled logit-space x
        xs[row * kD + k] = v;
        unsigned short h = f32_to_bf16_rne(v);
        float hf = bf16_to_f32(h);
        unsigned short l = f32_to_bf16_rne(v - hf);   // exact residual split
        xhi[row * kD + k] = h;
        xlo[row * kD + k] = l;
    }
    if (t == 0) rowsum[row] = 0.f;
}

// =====================================================================
// Kernel 2: 128x128 tile GEMM (split-bf16, 3 WMMAs per K-chunk per tile)
// fused with exp() and per-row sum-exp accumulation via f32 atomics.
// 256 threads = 8 waves arranged 4(M) x 2(N); wave tile = 32x64.
// B (features) staged through double-buffered LDS in WMMA-B layout.
// =====================================================================
__global__ void __launch_bounds__(256, 1)
gemm_sumexp_kernel(const float* __restrict__ features,
                   const unsigned short* __restrict__ xhi,
                   const unsigned short* __restrict__ xlo,
                   float* __restrict__ rowsum) {
    __shared__ __align__(16) unsigned short BhiS[2][BN][PITCH];
    __shared__ __align__(16) unsigned short BloS[2][BN][PITCH];

    const int tid  = threadIdx.x;
    const int lane = tid & 31;
    const int w    = tid >> 5;
    const int wm   = w & 3;          // 0..3 -> M offset
    const int wn   = w >> 2;         // 0..1 -> N offset
    const int l15  = lane & 15;
    const int lhi  = lane >> 4;      // 0/1 (lane half)

    const int nbase = blockIdx.x * BN;
    const int mbase = blockIdx.y * BM;

    // staging mapping: each thread loads 4x float4 (vectorized along N)
    const int sn4 = (tid & 31) * 4;  // n offset 0..124
    const int skb = tid >> 5;        // k base 0..7

    v8f acc[2][4];
#pragma unroll
    for (int mt = 0; mt < 2; ++mt)
#pragma unroll
        for (int nt = 0; nt < 4; ++nt)
#pragma unroll
            for (int r = 0; r < 8; ++r) acc[mt][nt][r] = 0.f;

    float4 fv[4];
    const bool ninb = (nbase + sn4) < kN;    // kN % 4 == 0, float4-safe guard

    // ---- prologue: load + stage K-chunk 0 into buffer 0 ----
#pragma unroll
    for (int j = 0; j < 4; ++j) {
        const int k = skb + 8 * j;
        fv[j] = ninb ? *(const float4*)(features + (size_t)k * kN + nbase + sn4)
                     : make_float4(0.f, 0.f, 0.f, 0.f);
    }
#pragma unroll
    for (int j = 0; j < 4; ++j) {
        const int k = skb + 8 * j;
        const float e[4] = {fv[j].x, fv[j].y, fv[j].z, fv[j].w};
#pragma unroll
        for (int c = 0; c < 4; ++c) {
            unsigned short h = f32_to_bf16_rne(e[c]);
            unsigned short l = f32_to_bf16_rne(e[c] - bf16_to_f32(h));
            BhiS[0][sn4 + c][k] = h;
            BloS[0][sn4 + c][k] = l;
        }
    }
    __syncthreads();

    // ---- main K loop (double-buffered) ----
    for (int kc = 0; kc < kD; kc += KC) {
        const int  buf     = (kc >> 5) & 1;
        const bool hasNext = (kc + KC) < kD;

        if (hasNext) {           // issue next chunk's global loads early
#pragma unroll
            for (int j = 0; j < 4; ++j) {
                const int k = kc + KC + skb + 8 * j;
                fv[j] = ninb ? *(const float4*)(features + (size_t)k * kN + nbase + sn4)
                             : make_float4(0.f, 0.f, 0.f, 0.f);
            }
        }

        // A operands: per-lane layout = two contiguous 8-half runs
        V16 ahi[2], alo[2];
#pragma unroll
        for (int mt = 0; mt < 2; ++mt) {
            const int m = mbase + wm * 32 + mt * 16 + l15;
            const size_t base = (size_t)m * kD + kc + lhi * 8;
            ahi[mt].u[0] = *(const us8*)(xhi + base);
            ahi[mt].u[1] = *(const us8*)(xhi + base + 16);
            alo[mt].u[0] = *(const us8*)(xlo + base);
            alo[mt].u[1] = *(const us8*)(xlo + base + 16);
        }

#pragma unroll
        for (int nt = 0; nt < 4; ++nt) {
            const int nloc = wn * 64 + nt * 16 + l15;   // lane = column
            const unsigned short* ph = &BhiS[buf][nloc][lhi * 8];
            const unsigned short* pl = &BloS[buf][nloc][lhi * 8];
            V16 bhi, blo;
            bhi.u[0] = *(const us8*)(ph);
            bhi.u[1] = *(const us8*)(ph + 16);
            blo.u[0] = *(const us8*)(pl);
            blo.u[1] = *(const us8*)(pl + 16);
#pragma unroll
            for (int mt = 0; mt < 2; ++mt) {
                acc[mt][nt] = __builtin_amdgcn_wmma_f32_16x16x32_bf16(
                    false, ahi[mt].bf, false, bhi.bf, (short)0, acc[mt][nt], false, false);
                acc[mt][nt] = __builtin_amdgcn_wmma_f32_16x16x32_bf16(
                    false, ahi[mt].bf, false, blo.bf, (short)0, acc[mt][nt], false, false);
                acc[mt][nt] = __builtin_amdgcn_wmma_f32_16x16x32_bf16(
                    false, alo[mt].bf, false, bhi.bf, (short)0, acc[mt][nt], false, false);
            }
        }

        if (hasNext) {           // convert + stage into the other buffer
            const int nb = buf ^ 1;
#pragma unroll
            for (int j = 0; j < 4; ++j) {
                const int k = skb + 8 * j;
                const float e[4] = {fv[j].x, fv[j].y, fv[j].z, fv[j].w};
#pragma unroll
                for (int c = 0; c < 4; ++c) {
                    unsigned short h = f32_to_bf16_rne(e[c]);
                    unsigned short l = f32_to_bf16_rne(e[c] - bf16_to_f32(h));
                    BhiS[nb][sn4 + c][k] = h;
                    BloS[nb][sn4 + c][k] = l;
                }
            }
        }
        __syncthreads();
    }

    // ---- epilogue: masked exp, 16-lane row reduce, atomic row accumulate ----
    const int ncolbase = nbase + wn * 64;
#pragma unroll
    for (int mt = 0; mt < 2; ++mt) {
#pragma unroll
        for (int r = 0; r < 8; ++r) {
            float p = 0.f;
#pragma unroll
            for (int nt = 0; nt < 4; ++nt) {
                const int n = ncolbase + nt * 16 + l15;
                const float v = acc[mt][nt][r];
                p += (n < kN) ? __expf(v) : 0.f;   // |v| <= 20: no overflow
            }
            p += __shfl_xor(p, 1, 32);
            p += __shfl_xor(p, 2, 32);
            p += __shfl_xor(p, 4, 32);
            p += __shfl_xor(p, 8, 32);
            if ((lane & 15) == 0) {
                const int m = mbase + wm * 32 + mt * 16 + r + lhi * 8;
                atomicAdd(&rowsum[m], p);
            }
        }
    }
}

// =====================================================================
// Kernel 3: exact f32 target logits: tgt[i] = xs[i] . features[:, t_i]
// =====================================================================
__global__ void tgt_kernel(const float* __restrict__ xs,
                           const float* __restrict__ features,
                           const int* __restrict__ targets,
                           float* __restrict__ tgt) {
    const int row = blockIdx.x;
    const int t   = threadIdx.x;
    const int cls = targets[row];
    __shared__ float red[256];

    float s = 0.f;
    for (int k = t; k < kD; k += 256)
        s += xs[row * kD + k] * features[(size_t)k * kN + cls];
    red[t] = s;
    __syncthreads();
    for (int o = 128; o > 0; o >>= 1) {
        if (t < o) red[t] += red[t + o];
        __syncthreads();
    }
    if (t == 0) tgt[row] = red[0];
}

// =====================================================================
// Kernel 4: loss = mean_i( log(sum_exp_i) - tgt_i )
// =====================================================================
__global__ void loss_kernel(const float* __restrict__ rowsum,
                            const float* __restrict__ tgt,
                            float* __restrict__ out) {
    const int t = threadIdx.x;
    __shared__ float red[256];
    red[t] = logf(rowsum[t]) - tgt[t];
    __syncthreads();
    for (int o = 128; o > 0; o >>= 1) {
        if (t < o) red[t] += red[t + o];
        __syncthreads();
    }
    if (t == 0) out[0] = red[0] * (1.0f / (float)kB);
}

// =====================================================================
extern "C" void kernel_launch(void* const* d_in, const int* in_sizes, int n_in,
                              void* d_out, int out_size, void* d_ws, size_t ws_size,
                              hipStream_t stream) {
    (void)in_sizes; (void)n_in; (void)out_size; (void)ws_size;
    const float* inputs   = (const float*)d_in[0];
    // d_in[1] = k_inputs (unused by forward)
    const int*   targets  = (const int*)d_in[2];
    const float* features = (const float*)d_in[3];

    char* ws = (char*)d_ws;
    const size_t nx = (size_t)kB * kD;
    float*          xs     = (float*)ws;                                  // 2 MB
    unsigned short* xhi    = (unsigned short*)(ws + nx * 4);              // 1 MB
    unsigned short* xlo    = (unsigned short*)(ws + nx * 4 + nx * 2);     // 1 MB
    float*          rowsum = (float*)(ws + nx * 8);                       // 1 KB
    float*          tgt    = rowsum + kB;                                 // 1 KB

    prep_kernel<<<kB, 256, 0, stream>>>(inputs, xs, xhi, xlo, rowsum);

    dim3 grid((kN + BN - 1) / BN, kB / BM);   // (782, 2)
    gemm_sumexp_kernel<<<grid, 256, 0, stream>>>(features, xhi, xlo, rowsum);

    tgt_kernel<<<kB, 256, 0, stream>>>(xs, features, targets, tgt);

    loss_kernel<<<1, 256, 0, stream>>>(rowsum, tgt, (float*)d_out);
}
